// SDHGCN_31937376813484
// MI455X (gfx1250) — matrix operand order repaired and checked by reading, not compiled, not run on hardware
//
#include <hip/hip_runtime.h>
#include <hip/hip_bf16.h>
#include <stdint.h>

#define N_NODES 8192
#define F_INF   512   // F_IN == F_OUT == 512

#define TILE_M 128
#define TILE_F 128
#define TILE_K 64
#define LDS_STRIDE (TILE_K + 8)   // 72 bf16 = 144 B rows: 16B-aligned, 36-dword stride

typedef __bf16 bf16x8  __attribute__((ext_vector_type(8)));
typedef __bf16 bf16x16 __attribute__((ext_vector_type(16)));
typedef float  f32x8   __attribute__((ext_vector_type(8)));

union FragAB { bf16x16 v; bf16x8 h[2]; };

// ---------------------------------------------------------------------------
// Kernel 1: XWT[f][m] = bf16( X[m][:] . W[:][f] )   (output stored transposed)
// ---------------------------------------------------------------------------
__global__ __launch_bounds__(256)
void gcn_xw_kernel(const float* __restrict__ X, const float* __restrict__ W,
                   __bf16* __restrict__ XWT) {
  __shared__ __align__(16) __bf16 Alds[TILE_M][LDS_STRIDE];
  __shared__ __align__(16) __bf16 Blds[TILE_F][LDS_STRIDE];

  const int t    = threadIdx.x;
  const int m0   = blockIdx.x * TILE_M;
  const int f0   = blockIdx.y * TILE_F;
  const int wave = t >> 5, lane = t & 31;
  const int m_wave = (wave & 3) * 32;
  const int f_wave = (wave >> 2) * 64;
  const int lhalf  = lane >> 4;        // 0..1
  const int lmod   = lane & 15;        // 0..15

  f32x8 acc[2][4];
#pragma unroll
  for (int mi = 0; mi < 2; mi++)
#pragma unroll
    for (int fi = 0; fi < 4; fi++)
#pragma unroll
      for (int e = 0; e < 8; e++) acc[mi][fi][e] = 0.0f;

  for (int k0 = 0; k0 < F_INF; k0 += TILE_K) {
    __syncthreads();
    // Stage A = X tile: X[m0+mm][k0+kk] -> Alds[mm][kk] (k contiguous in mem)
    {
      const int kseg = (t & 15) * 4;
      const int mrow = t >> 4;
#pragma unroll
      for (int i = 0; i < 8; i++) {
        const int mm = mrow + i * 16;
        float4 v = *(const float4*)(X + (size_t)(m0 + mm) * F_INF + k0 + kseg);
        __bf16* dst = &Alds[mm][kseg];
        dst[0] = (__bf16)v.x; dst[1] = (__bf16)v.y;
        dst[2] = (__bf16)v.z; dst[3] = (__bf16)v.w;
      }
    }
    // Stage B = W tile: W[k0+kk][f0+ff] -> Blds[ff][kk] (transpose scatter)
    {
      const int ff = (t & 31) * 4;
      const int kb = t >> 5;
#pragma unroll
      for (int i = 0; i < 8; i++) {
        const int kk = kb + i * 8;
        float4 v = *(const float4*)(W + (size_t)(k0 + kk) * F_INF + f0 + ff);
        Blds[ff + 0][kk] = (__bf16)v.x;
        Blds[ff + 1][kk] = (__bf16)v.y;
        Blds[ff + 2][kk] = (__bf16)v.z;
        Blds[ff + 3][kk] = (__bf16)v.w;
      }
    }
    __syncthreads();

#pragma unroll
    for (int ks = 0; ks < TILE_K; ks += 32) {
      FragAB a[2], b[4];
#pragma unroll
      for (int mi = 0; mi < 2; mi++) {
        const __bf16* p = &Alds[m_wave + mi * 16 + lmod][ks + lhalf * 8];
        a[mi].h[0] = *(const bf16x8*)p;          // K {0-7} / {8-15}
        a[mi].h[1] = *(const bf16x8*)(p + 16);   // K {16-23} / {24-31}
      }
#pragma unroll
      for (int fi = 0; fi < 4; fi++) {
        const __bf16* p = &Blds[f_wave + fi * 16 + lmod][ks + lhalf * 8];
        b[fi].h[0] = *(const bf16x8*)p;          // K {0-7} / {8-15}
        b[fi].h[1] = *(const bf16x8*)(p + 16);   // K {16-23} / {24-31}
      }
#pragma unroll
      for (int mi = 0; mi < 2; mi++)
#pragma unroll
        for (int fi = 0; fi < 4; fi++)
          acc[mi][fi] = __builtin_amdgcn_wmma_f32_16x16x32_bf16(
              false, a[mi].v, false, b[fi].v, (short)0, acc[mi][fi], false, false);
    }
  }

  // Epilogue: write bf16, transposed: XWT[f][m]
#pragma unroll
  for (int mi = 0; mi < 2; mi++)
#pragma unroll
    for (int fi = 0; fi < 4; fi++) {
      const int f     = f0 + f_wave + fi * 16 + lmod;
      const int mbase = m0 + m_wave + mi * 16 + lhalf * 8;
#pragma unroll
      for (int r = 0; r < 8; r++)
        XWT[(size_t)f * N_NODES + mbase + r] = (__bf16)acc[mi][fi][r];
    }
}

// ---------------------------------------------------------------------------
// Kernel 2: out[m][f] = relu( rsqrt(max(deg[m],1)) * sum_k adj[k][m]*XWT[f][k] )
// Degree fused into the adjacency staging loop (free: same loads).
// B tile staged with GLOBAL_LOAD_ASYNC_TO_LDS_B64 (ASYNCcnt path): the bf16
// copy bypasses VGPRs and overlaps with the int32->bf16 adjacency conversion.
// ---------------------------------------------------------------------------
__global__ __launch_bounds__(256)
void gcn_agg_kernel(const int* __restrict__ adj, const __bf16* __restrict__ XWT,
                    float* __restrict__ out) {
  __shared__ __align__(16) __bf16 Alds[TILE_M][LDS_STRIDE];
  __shared__ __align__(16) __bf16 Blds[TILE_F][LDS_STRIDE];
  __shared__ int   degLds[TILE_M];
  __shared__ float normLds[TILE_M];

  const int t    = threadIdx.x;
  const int m0   = blockIdx.x * TILE_M;   // output rows = adj columns
  const int f0   = blockIdx.y * TILE_F;
  const int wave = t >> 5, lane = t & 31;
  const int m_wave = (wave & 3) * 32;
  const int f_wave = (wave >> 2) * 64;
  const int lhalf  = lane >> 4;
  const int lmod   = lane & 15;

  const int mseg = (t & 31) * 4;   // 4 adj columns owned by this thread
  const int kb   = t >> 5;

  f32x8 acc[2][4];
#pragma unroll
  for (int mi = 0; mi < 2; mi++)
#pragma unroll
    for (int fi = 0; fi < 4; fi++)
#pragma unroll
      for (int e = 0; e < 8; e++) acc[mi][fi][e] = 0.0f;

  int degAcc[4] = {0, 0, 0, 0};

  // Per-thread B-staging geometry: 4 bf16 (8 bytes) per row, 8 rows per stage
  const int kseg = (t & 15) * 4;
  const int fr   = t >> 4;

  for (int k0 = 0; k0 < N_NODES; k0 += TILE_K) {
    __syncthreads();   // all waves done reading previous LDS tiles

    // --- Issue async B-tile copies first (LDS write, no VGPR data path) ---
#pragma unroll
    for (int i = 0; i < 8; i++) {
      const int ff = fr + i * 16;
      const uint32_t lds_off =
          (uint32_t)(uintptr_t)(&Blds[ff][kseg]);           // LDS byte address
      const uint64_t gaddr =
          (uint64_t)(uintptr_t)(XWT + (size_t)(f0 + ff) * N_NODES + k0 + kseg);
      asm volatile("global_load_async_to_lds_b64 %0, %1, off"
                   :: "v"(lds_off), "v"(gaddr) : "memory");
    }

    // --- Stage A = adj tile: adj[k0+kk][m0+mseg..+3] (coalesced int4),
    //     transpose-scatter to Alds[mm][kk] as bf16; accumulate in-degree. ---
#pragma unroll
    for (int i = 0; i < 8; i++) {
      const int kk = kb + i * 8;
      const int* gp = adj + (size_t)(k0 + kk) * N_NODES + m0 + mseg;
      if (k0 + TILE_K < N_NODES)
        __builtin_prefetch(gp + (size_t)TILE_K * N_NODES, 0, 1);
      int4 v = *(const int4*)gp;
      degAcc[0] += v.x; degAcc[1] += v.y; degAcc[2] += v.z; degAcc[3] += v.w;
      Alds[mseg + 0][kk] = (__bf16)(float)v.x;
      Alds[mseg + 1][kk] = (__bf16)(float)v.y;
      Alds[mseg + 2][kk] = (__bf16)(float)v.z;
      Alds[mseg + 3][kk] = (__bf16)(float)v.w;
    }

    asm volatile("s_wait_asynccnt 0x0" ::: "memory");   // B tile landed in LDS
    __syncthreads();

#pragma unroll
    for (int ks = 0; ks < TILE_K; ks += 32) {
      FragAB a[2], b[4];
#pragma unroll
      for (int mi = 0; mi < 2; mi++) {
        const __bf16* p = &Alds[m_wave + mi * 16 + lmod][ks + lhalf * 8];
        a[mi].h[0] = *(const bf16x8*)p;
        a[mi].h[1] = *(const bf16x8*)(p + 16);
      }
#pragma unroll
      for (int fi = 0; fi < 4; fi++) {
        const __bf16* p = &Blds[f_wave + fi * 16 + lmod][ks + lhalf * 8];
        b[fi].h[0] = *(const bf16x8*)p;
        b[fi].h[1] = *(const bf16x8*)(p + 16);
      }
#pragma unroll
      for (int mi = 0; mi < 2; mi++)
#pragma unroll
        for (int fi = 0; fi < 4; fi++)
          acc[mi][fi] = __builtin_amdgcn_wmma_f32_16x16x32_bf16(
              false, a[mi].v, false, b[fi].v, (short)0, acc[mi][fi], false, false);
    }
  }

  // Degree reduction -> norm (redundant per F-block; avoids cross-WG sync)
  __syncthreads();
  if (t < TILE_M) degLds[t] = 0;
  __syncthreads();
  atomicAdd(&degLds[mseg + 0], degAcc[0]);
  atomicAdd(&degLds[mseg + 1], degAcc[1]);
  atomicAdd(&degLds[mseg + 2], degAcc[2]);
  atomicAdd(&degLds[mseg + 3], degAcc[3]);
  __syncthreads();
  if (t < TILE_M)
    normLds[t] = rsqrtf(fmaxf((float)degLds[t], 1.0f));
  __syncthreads();

  // Epilogue: scale + relu, store f32
#pragma unroll
  for (int mi = 0; mi < 2; mi++)
#pragma unroll
    for (int fi = 0; fi < 4; fi++) {
      const int f    = f0 + f_wave + fi * 16 + lmod;
      const int mloc = m_wave + mi * 16 + lhalf * 8;
#pragma unroll
      for (int r = 0; r < 8; r++) {
        float v = acc[mi][fi][r] * normLds[mloc + r];
        v = fmaxf(v, 0.0f);
        out[(size_t)(m0 + mloc + r) * F_INF + f] = v;
      }
    }
}

// ---------------------------------------------------------------------------
extern "C" void kernel_launch(void* const* d_in, const int* in_sizes, int n_in,
                              void* d_out, int out_size, void* d_ws, size_t ws_size,
                              hipStream_t stream) {
  const float* X   = (const float*)d_in[0];
  const int*   adj = (const int*)d_in[1];
  const float* W   = (const float*)d_in[2];
  float*  out = (float*)d_out;
  __bf16* XWT = (__bf16*)d_ws;   // 8192*512*2 = 8 MB scratch, transposed XW

  dim3 block(256);
  dim3 grid(N_NODES / TILE_M, F_INF / TILE_F);   // 64 x 4

  gcn_xw_kernel<<<grid, block, 0, stream>>>(X, W, XWT);
  gcn_agg_kernel<<<grid, block, 0, stream>>>(adj, XWT, out);
}
